// TripleStreamBlock_59725815218900
// MI455X (gfx1250) — matrix operand order repaired and checked by reading, not compiled, not run on hardware
//
#include <hip/hip_runtime.h>
#include <hip/hip_bf16.h>

// ---------------------------------------------------------------------------
// MI455X (gfx1250) triple-stream DiT block.
// Compute-bound (~80 GFLOP vs ~240MB traffic @ 23.3TB/s): run every GEMM and
// the attention on v_wmma_f32_16x16x32_f16 (f16 in, f32 accum), converting
// fp32 -> f16 while staging tiles in LDS. wave32, 8 waves / 256-thread block.
// ---------------------------------------------------------------------------

typedef __attribute__((ext_vector_type(16))) _Float16 v16h;
typedef __attribute__((ext_vector_type(8)))  _Float16 v8h;
typedef __attribute__((ext_vector_type(8)))  float    v8f;

#define LBAR 72            // LDS row pitch in halves (64 + 8 pad)
#define L_TOT 2304

__device__ __forceinline__ v16h mkfrag(v8h lo, v8h hi) {
  v16h r;
#pragma unroll
  for (int i = 0; i < 8; ++i) { r[i] = lo[i]; r[i + 8] = hi[i]; }
  return r;
}

__device__ __forceinline__ float gelu_tanh(float x) {
  float x3 = x * x * x;
  return 0.5f * x * (1.0f + tanhf(0.7978845608028654f * (x + 0.044715f * x3)));
}

// ---------------------------------------------------------------------------
// Tiled WMMA GEMM: C(MxN) = A(MxK, f16 row-major) * B(KxN, f32 row-major)
// Block tile 128x128, K-step 64. 8 waves arranged 4(M) x 2(N); each wave owns
// a 32x64 tile = 2x4 wmma accumulators.
// EPI 0: raw f32 store              (qkv)
// EPI 1: +bias, *gate, +residual    (proj / mlp2)  -> f32
// EPI 2: +bias, gelu                (mlp1)         -> f16
// ---------------------------------------------------------------------------
template <int EPI>
__global__ __launch_bounds__(256) void gemm_kernel(
    const _Float16* __restrict__ A, int lda,
    const float* __restrict__ B, int ldb,
    int M, int N, int K,
    const float* __restrict__ bias,
    const float* __restrict__ gate,
    const float* __restrict__ resid,
    float* __restrict__ outf,
    _Float16* __restrict__ outh) {
  __shared__ __align__(16) _Float16 sA[128 * LBAR];
  __shared__ __align__(16) _Float16 sB[128 * LBAR];   // transposed: [n][k]

  const int t = threadIdx.x, lane = t & 31, wave = t >> 5;
  const int m0 = blockIdx.y * 128, n0 = blockIdx.x * 128;
  const int wm = (wave & 3) * 32, wn = (wave >> 2) * 64;
  const int half = (lane & 16) ? 1 : 0;

  v8f acc[2][4];
#pragma unroll
  for (int mi = 0; mi < 2; ++mi)
#pragma unroll
    for (int ni = 0; ni < 4; ++ni)
#pragma unroll
      for (int r = 0; r < 8; ++r) acc[mi][ni][r] = 0.0f;

  for (int k0 = 0; k0 < K; k0 += 64) {
    __syncthreads();
    // stage A: 128 rows x 64 halves (f16 already), 4 x b128 per thread
#pragma unroll
    for (int l = 0; l < 4; ++l) {
      int c = t + l * 256, r = c >> 3, s = c & 7;
      *(v8h*)&sA[r * LBAR + s * 8] =
          *(const v8h*)&A[(size_t)(m0 + r) * lda + k0 + s * 8];
    }
    // stage B transposed with f32->f16 convert: 64 k x 128 n floats
#pragma unroll
    for (int l = 0; l < 8; ++l) {
      int c = t + l * 256, kk = c >> 5, n4 = (c & 31) << 2;
      const float4 f = *(const float4*)&B[(size_t)(k0 + kk) * ldb + n0 + n4];
      sB[(n4 + 0) * LBAR + kk] = (_Float16)f.x;
      sB[(n4 + 1) * LBAR + kk] = (_Float16)f.y;
      sB[(n4 + 2) * LBAR + kk] = (_Float16)f.z;
      sB[(n4 + 3) * LBAR + kk] = (_Float16)f.w;
    }
    if (k0 + 64 < K)  // pull next weight tile toward GL2 (global_prefetch_b8)
      __builtin_prefetch(&B[(size_t)(k0 + 64) * ldb + n0 + ((t & 63) << 1)], 0, 1);
    __syncthreads();

#pragma unroll
    for (int ks = 0; ks < 2; ++ks) {
      const int kb = ks * 32;
      v16h af[2], bf[4];
      const int aoff = kb + (half ? 8 : 0);
#pragma unroll
      for (int mi = 0; mi < 2; ++mi) {
        const _Float16* p = &sA[(wm + mi * 16 + (lane & 15)) * LBAR + aoff];
        af[mi] = mkfrag(*(const v8h*)p, *(const v8h*)(p + 16));
      }
      const int boff = kb + (half ? 8 : 0);
#pragma unroll
      for (int ni = 0; ni < 4; ++ni) {
        const _Float16* p = &sB[(wn + ni * 16 + (lane & 15)) * LBAR + boff];
        bf[ni] = mkfrag(*(const v8h*)p, *(const v8h*)(p + 16));
      }
#pragma unroll
      for (int mi = 0; mi < 2; ++mi)
#pragma unroll
        for (int ni = 0; ni < 4; ++ni)
          acc[mi][ni] = __builtin_amdgcn_wmma_f32_16x16x32_f16(
              false, af[mi], false, bf[ni], (short)0, acc[mi][ni], false, false);
    }
  }

  // epilogue: C-layout -> lane<16 slot r = (row r, col lane); lane>=16 row r+8
#pragma unroll
  for (int mi = 0; mi < 2; ++mi)
#pragma unroll
    for (int ni = 0; ni < 4; ++ni)
#pragma unroll
      for (int r = 0; r < 8; ++r) {
        const int row = m0 + wm + mi * 16 + r + (half ? 8 : 0);
        const int col = n0 + wn + ni * 16 + (lane & 15);
        float c = acc[mi][ni][r];
        if (EPI == 0) {
          outf[(size_t)row * N + col] = c;
        } else if (EPI == 1) {
          c += bias[col];
          outf[(size_t)row * N + col] =
              resid[(size_t)row * N + col] + gate[col] * c;
        } else {  // EPI == 2
          c += bias[col];
          outh[(size_t)row * N + col] = (_Float16)gelu_tanh(c);
        }
      }
}

// ---------------------------------------------------------------------------
// silu(vec) and tiny modulation GEMV: mods[3][6*1024]
// ---------------------------------------------------------------------------
__global__ void silu_kernel(const float* __restrict__ v, float* __restrict__ sv) {
  int i = blockIdx.x * 256 + threadIdx.x;
  if (i < 1024) { float x = v[i]; sv[i] = x / (1.0f + __expf(-x)); }
}

__global__ void mods_kernel(const float* __restrict__ sv,
                            const float* __restrict__ w,
                            const float* __restrict__ b,
                            float* __restrict__ mods) {
  int i = blockIdx.x * 256 + threadIdx.x;      // 0 .. 3*6144-1
  int s = i / 6144, n = i - s * 6144;
  const float* wc = w + (size_t)s * 1024 * 6144 + n;
  float acc = b[i];
  for (int k = 0; k < 1024; ++k) acc += sv[k] * wc[(size_t)k * 6144];
  mods[i] = acc;
}

// ---------------------------------------------------------------------------
// Per-row LayerNorm + modulation: out = (1+sc)*ln(x) + sh  -> f16
// ---------------------------------------------------------------------------
__global__ __launch_bounds__(256) void ln_mod_kernel(
    const float* __restrict__ x, const float* __restrict__ sh,
    const float* __restrict__ sc, _Float16* __restrict__ out) {
  __shared__ float rb[16];
  const int t = threadIdx.x, lane = t & 31, wave = t >> 5;
  const float* xr = x + (size_t)blockIdx.x * 1024;
  float s = 0.f, s2 = 0.f;
  for (int i = t; i < 1024; i += 256) { float v = xr[i]; s += v; s2 += v * v; }
#pragma unroll
  for (int m = 1; m < 32; m <<= 1) {
    s += __shfl_xor(s, m, 32);
    s2 += __shfl_xor(s2, m, 32);
  }
  if (lane == 0) { rb[wave] = s; rb[8 + wave] = s2; }
  __syncthreads();
  float S = 0.f, S2 = 0.f;
#pragma unroll
  for (int i = 0; i < 8; ++i) { S += rb[i]; S2 += rb[8 + i]; }
  const float mean = S * (1.0f / 1024.0f);
  const float var = S2 * (1.0f / 1024.0f) - mean * mean;
  const float inv = rsqrtf(var + 1e-6f);
  _Float16* orow = out + (size_t)blockIdx.x * 1024;
  for (int i = t; i < 1024; i += 256)
    orow[i] = (_Float16)((1.0f + sc[i]) * ((xr[i] - mean) * inv) + sh[i]);
}

// ---------------------------------------------------------------------------
// RMS-norm q/k + RoPE + pack to (H, L, HD) f16. One wave per (pos, head).
// Concat order: txt [0,256), img [256,1280), tmp [1280,2304).
// Reference quirk: v for the tmp segment = rms'd tmp_q (pre-RoPE).
// ---------------------------------------------------------------------------
__global__ __launch_bounds__(256) void qkv_post_kernel(
    const float* __restrict__ qkv, const float* __restrict__ qs,
    const float* __restrict__ ksc, const float* __restrict__ pe,
    _Float16* __restrict__ qp, _Float16* __restrict__ kp,
    _Float16* __restrict__ vp) {
  const int pair = blockIdx.x * 8 + (threadIdx.x >> 5);
  const int lane = threadIdx.x & 31;
  const int pos = pair >> 4, h = pair & 15;
  const int stream = (pos < 256) ? 2 : ((pos < 1280) ? 0 : 1);
  const float* row = qkv + (size_t)pos * 3072 + h * 64;
  const int d0 = lane * 2;

  float q0 = row[d0], q1 = row[d0 + 1];
  float k0 = row[1024 + d0], k1 = row[1024 + d0 + 1];
  float v0 = row[2048 + d0], v1 = row[2048 + d0 + 1];

  float qss = q0 * q0 + q1 * q1, kss = k0 * k0 + k1 * k1;
#pragma unroll
  for (int m = 1; m < 32; m <<= 1) {
    qss += __shfl_xor(qss, m, 32);
    kss += __shfl_xor(kss, m, 32);
  }
  const float qinv = rsqrtf(qss * (1.0f / 64.0f) + 1e-6f);
  const float kinv = rsqrtf(kss * (1.0f / 64.0f) + 1e-6f);
  q0 *= qinv * qs[stream * 64 + d0];
  q1 *= qinv * qs[stream * 64 + d0 + 1];
  k0 *= kinv * ksc[stream * 64 + d0];
  k1 *= kinv * ksc[stream * 64 + d0 + 1];

  const float* p = pe + ((size_t)pos * 32 + lane) * 4;  // (L, 32, 2, 2)
  const float qo0 = p[0] * q0 + p[1] * q1, qo1 = p[2] * q0 + p[3] * q1;
  const float ko0 = p[0] * k0 + p[1] * k1, ko1 = p[2] * k0 + p[3] * k1;

  const size_t o = ((size_t)h * L_TOT + pos) * 64 + d0;
  qp[o] = (_Float16)qo0; qp[o + 1] = (_Float16)qo1;
  kp[o] = (_Float16)ko0; kp[o + 1] = (_Float16)ko1;
  if (stream == 1) { vp[o] = (_Float16)q0; vp[o + 1] = (_Float16)q1; }
  else             { vp[o] = (_Float16)v0; vp[o + 1] = (_Float16)v1; }
}

// ---------------------------------------------------------------------------
// Flash attention: grid (18 q-blocks, 16 heads), 8 waves x 16 q-rows each.
// K tile + transposed-V tile in LDS; per-wave P restage through LDS.
// ---------------------------------------------------------------------------
__global__ __launch_bounds__(256) void flash_kernel(
    const _Float16* __restrict__ qpk, const _Float16* __restrict__ kpk,
    const _Float16* __restrict__ vpk, _Float16* __restrict__ attn) {
  __shared__ __align__(16) _Float16 sK[64 * LBAR];
  __shared__ __align__(16) _Float16 sVT[64 * LBAR];
  __shared__ __align__(16) _Float16 sP[8 * 16 * LBAR];

  const int t = threadIdx.x, lane = t & 31, wave = t >> 5;
  const int qb = blockIdx.x, h = blockIdx.y;
  const int qrow0 = qb * 128 + wave * 16;
  const int half = (lane & 16) ? 1 : 0;
  const int aoff = half ? 8 : 0;

  // Q fragments held in registers for the whole kernel (K-dim = HD = 64)
  const _Float16* qrowp = qpk + ((size_t)h * L_TOT + qrow0 + (lane & 15)) * 64;
  v16h qf[2];
  qf[0] = mkfrag(*(const v8h*)(qrowp + aoff), *(const v8h*)(qrowp + aoff + 16));
  qf[1] = mkfrag(*(const v8h*)(qrowp + 32 + aoff),
                 *(const v8h*)(qrowp + 32 + aoff + 16));

  float mi[8], li[8];
  v8f o[4];
#pragma unroll
  for (int r = 0; r < 8; ++r) { mi[r] = -1e30f; li[r] = 0.0f; }
#pragma unroll
  for (int d = 0; d < 4; ++d)
#pragma unroll
    for (int r = 0; r < 8; ++r) o[d][r] = 0.0f;

  _Float16* pp = sP + wave * 16 * LBAR;

  for (int kb = 0; kb < 36; ++kb) {
    __syncthreads();
#pragma unroll
    for (int l = 0; l < 2; ++l) {
      int c = t + l * 256, key = c >> 3, s = c & 7;
      *(v8h*)&sK[key * LBAR + s * 8] =
          *(const v8h*)&kpk[((size_t)h * L_TOT + kb * 64 + key) * 64 + s * 8];
      v8h tv =
          *(const v8h*)&vpk[((size_t)h * L_TOT + kb * 64 + key) * 64 + s * 8];
#pragma unroll
      for (int i = 0; i < 8; ++i) sVT[(s * 8 + i) * LBAR + key] = tv[i];
    }
    __syncthreads();

    // S = Q @ K^T (16 x 64 per wave)
    v8f sa[4];
#pragma unroll
    for (int nt = 0; nt < 4; ++nt)
#pragma unroll
      for (int r = 0; r < 8; ++r) sa[nt][r] = 0.0f;
#pragma unroll
    for (int kc = 0; kc < 2; ++kc) {
      const int boff = kc * 32 + (half ? 8 : 0);
#pragma unroll
      for (int nt = 0; nt < 4; ++nt) {
        const _Float16* p = &sK[(nt * 16 + (lane & 15)) * LBAR + boff];
        v16h bf = mkfrag(*(const v8h*)p, *(const v8h*)(p + 16));
        sa[nt] = __builtin_amdgcn_wmma_f32_16x16x32_f16(
            false, qf[kc], false, bf, (short)0, sa[nt], false, false);
      }
    }
#pragma unroll
    for (int nt = 0; nt < 4; ++nt)
#pragma unroll
      for (int r = 0; r < 8; ++r) sa[nt][r] *= 0.125f;  // 1/sqrt(HD)

    // online softmax (row reductions across the 16 lanes of each half)
    float mx[8], al[8], rs[8];
#pragma unroll
    for (int r = 0; r < 8; ++r) {
      float v = fmaxf(fmaxf(sa[0][r], sa[1][r]), fmaxf(sa[2][r], sa[3][r]));
#pragma unroll
      for (int m = 1; m < 16; m <<= 1) v = fmaxf(v, __shfl_xor(v, m, 32));
      mx[r] = v;
      float mn = fmaxf(mi[r], v);
      al[r] = __expf(mi[r] - mn);
      mi[r] = mn;
      rs[r] = 0.0f;
    }
#pragma unroll
    for (int nt = 0; nt < 4; ++nt)
#pragma unroll
      for (int r = 0; r < 8; ++r) {
        float pv = __expf(sa[nt][r] - mi[r]);
        sa[nt][r] = pv;
        rs[r] += pv;
      }
#pragma unroll
    for (int r = 0; r < 8; ++r) {
#pragma unroll
      for (int m = 1; m < 16; m <<= 1) rs[r] += __shfl_xor(rs[r], m, 32);
      li[r] = li[r] * al[r] + rs[r];
    }
#pragma unroll
    for (int d = 0; d < 4; ++d)
#pragma unroll
      for (int r = 0; r < 8; ++r) o[d][r] *= al[r];

    // restage P (C-layout) into A-fragment layout via per-wave LDS
    const int pro = half ? 8 : 0;
#pragma unroll
    for (int nt = 0; nt < 4; ++nt)
#pragma unroll
      for (int r = 0; r < 8; ++r)
        pp[(r + pro) * LBAR + nt * 16 + (lane & 15)] = (_Float16)sa[nt][r];

    // O += P @ V
#pragma unroll
    for (int kc = 0; kc < 2; ++kc) {
      const _Float16* ap = pp + (lane & 15) * LBAR + kc * 32 + aoff;
      v16h pf = mkfrag(*(const v8h*)ap, *(const v8h*)(ap + 16));
      const int boff = kc * 32 + (half ? 8 : 0);
#pragma unroll
      for (int dt = 0; dt < 4; ++dt) {
        const _Float16* bp = &sVT[(dt * 16 + (lane & 15)) * LBAR + boff];
        v16h vf = mkfrag(*(const v8h*)bp, *(const v8h*)(bp + 16));
        o[dt] = __builtin_amdgcn_wmma_f32_16x16x32_f16(
            false, pf, false, vf, (short)0, o[dt], false, false);
      }
    }
  }

  // write attn (L, H*HD) f16, head-major columns
#pragma unroll
  for (int dt = 0; dt < 4; ++dt)
#pragma unroll
    for (int r = 0; r < 8; ++r) {
      const int row = qrow0 + r + (half ? 8 : 0);
      const int col = h * 64 + dt * 16 + (lane & 15);
      attn[(size_t)row * 1024 + col] = (_Float16)(o[dt][r] / li[r]);
    }
}

// ---------------------------------------------------------------------------
// Host side
// ---------------------------------------------------------------------------
extern "C" void kernel_launch(void* const* d_in, const int* in_sizes, int n_in,
                              void* d_out, int out_size, void* d_ws,
                              size_t ws_size, hipStream_t stream) {
  const float* img      = (const float*)d_in[0];
  const float* temporal = (const float*)d_in[1];
  const float* txt      = (const float*)d_in[2];
  const float* vec      = (const float*)d_in[3];
  const float* pe       = (const float*)d_in[4];
  const float* mod_w    = (const float*)d_in[5];
  const float* mod_b    = (const float*)d_in[6];
  const float* qkv_w    = (const float*)d_in[7];
  const float* q_scale  = (const float*)d_in[8];
  const float* k_scale  = (const float*)d_in[9];
  const float* proj_w   = (const float*)d_in[10];
  const float* proj_b   = (const float*)d_in[11];
  const float* mlp_w1   = (const float*)d_in[12];
  const float* mlp_b1   = (const float*)d_in[13];
  const float* mlp_w2   = (const float*)d_in[14];
  const float* mlp_b2   = (const float*)d_in[15];

  float* out     = (float*)d_out;
  float* out_img = out;                       // (1024, 1024)
  float* out_tmp = out + 1024 * 1024;         // (1024, 1024)
  float* out_txt = out + 2 * 1024 * 1024;     // (256, 1024)

  char* w = (char*)d_ws;
  auto alloc = [&](size_t bytes) -> void* {
    void* p = (void*)w;
    w += (bytes + 255) & ~(size_t)255;
    return p;
  };
  float*     sv     = (float*)alloc(1024 * 4);
  float*     mods   = (float*)alloc(3 * 6144 * 4);
  _Float16*  xm1    = (_Float16*)alloc((size_t)2304 * 1024 * 2);
  float*     qkvraw = (float*)alloc((size_t)2304 * 3072 * 4);
  _Float16*  qpk    = (_Float16*)alloc((size_t)16 * 2304 * 64 * 2);
  _Float16*  kpk    = (_Float16*)alloc((size_t)16 * 2304 * 64 * 2);
  _Float16*  vpk    = (_Float16*)alloc((size_t)16 * 2304 * 64 * 2);
  _Float16*  attn   = (_Float16*)alloc((size_t)2304 * 1024 * 2);
  _Float16*  xm2    = (_Float16*)alloc((size_t)1024 * 1024 * 2);
  _Float16*  hbuf   = (_Float16*)alloc((size_t)1024 * 4096 * 2);

  silu_kernel<<<4, 256, 0, stream>>>(vec, sv);
  mods_kernel<<<72, 256, 0, stream>>>(sv, mod_w, mod_b, mods);

  struct Str { const float* xin; float* xout; int sidx; int rowoff; int rows; };
  const Str st[3] = {
      {txt,      out_txt, 2, 0,    256},
      {img,      out_img, 0, 256,  1024},
      {temporal, out_tmp, 1, 1280, 1024},
  };

  // LN1 + modulate, QKV GEMM per stream (concat layout)
  for (int i = 0; i < 3; ++i) {
    const Str& s = st[i];
    ln_mod_kernel<<<s.rows, 256, 0, stream>>>(
        s.xin, mods + s.sidx * 6144 + 0 * 1024, mods + s.sidx * 6144 + 1 * 1024,
        xm1 + (size_t)s.rowoff * 1024);
    gemm_kernel<0><<<dim3(3072 / 128, s.rows / 128), 256, 0, stream>>>(
        xm1 + (size_t)s.rowoff * 1024, 1024,
        qkv_w + (size_t)s.sidx * 1024 * 3072, 3072,
        s.rows, 3072, 1024, nullptr, nullptr, nullptr,
        qkvraw + (size_t)s.rowoff * 3072, nullptr);
  }

  qkv_post_kernel<<<4608, 256, 0, stream>>>(qkvraw, q_scale, k_scale, pe,
                                            qpk, kpk, vpk);
  flash_kernel<<<dim3(18, 16), 256, 0, stream>>>(qpk, kpk, vpk, attn);

  // proj + residual, LN2 + modulate, MLP. Order: img first (temporal's LN2
  // reads the *final* img stream, per the reference), then temporal, then txt.
  auto tail = [&](const Str& s, const float* lnsrc) {
    gemm_kernel<1><<<dim3(1024 / 128, s.rows / 128), 256, 0, stream>>>(
        attn + (size_t)s.rowoff * 1024, 1024,
        proj_w + (size_t)s.sidx * 1024 * 1024, 1024,
        s.rows, 1024, 1024, proj_b + s.sidx * 1024,
        mods + s.sidx * 6144 + 2 * 1024, s.xin, s.xout, nullptr);
    ln_mod_kernel<<<s.rows, 256, 0, stream>>>(
        lnsrc, mods + s.sidx * 6144 + 3 * 1024,
        mods + s.sidx * 6144 + 4 * 1024, xm2);
    gemm_kernel<2><<<dim3(4096 / 128, s.rows / 128), 256, 0, stream>>>(
        xm2, 1024, mlp_w1 + (size_t)s.sidx * 1024 * 4096, 4096,
        s.rows, 4096, 1024, mlp_b1 + s.sidx * 4096,
        nullptr, nullptr, nullptr, hbuf);
    gemm_kernel<1><<<dim3(1024 / 128, s.rows / 128), 256, 0, stream>>>(
        hbuf, 4096, mlp_w2 + (size_t)s.sidx * 4096 * 1024, 1024,
        s.rows, 1024, 4096, mlp_b2 + s.sidx * 1024,
        mods + s.sidx * 6144 + 5 * 1024, s.xout, s.xout, nullptr);
  };

  tail(st[1], out_img);   // img    : LN2 over updated img
  tail(st[2], out_img);   // temporal: LN2 over FINAL img (reference quirk)
  tail(st[0], out_txt);   // txt    : LN2 over updated txt
}
